// Network_30193620091318
// MI455X (gfx1250) — matrix operand order repaired and checked by reading, not compiled, not run on hardware
//
#include <hip/hip_runtime.h>
#include <hip/hip_bf16.h>
#include <math.h>

typedef __attribute__((ext_vector_type(16))) _Float16 v16h;
typedef __attribute__((ext_vector_type(8)))  _Float16 v8h;
typedef __attribute__((ext_vector_type(8)))  float    v8f;

#define SQH   0.70710678118654752f   // sqrt(0.5)
#define SH0c  0.22360679774997896f   // 1/sqrt(20)
#define RS50  0.14142135623730950f   // 1/sqrt(50)
#define RS60  0.12909944487358056f   // 1/sqrt(60)
#define RS10  0.31622776601683794f   // 1/sqrt(10)
#define RS3   0.57735026918962576f   // 1/sqrt(3)
#define RS2   0.70710678118654752f   // 1/sqrt(2)

__device__ __forceinline__ float swish_f(float x) { return x / (1.f + __expf(-x)); }
__device__ __forceinline__ float sigm_f(float x)  { return 1.f / (1.f + __expf(-x)); }

// ---------------------------------------------------------------------------
// Pack an f32 weight matrix [K x N] (row-major) into padded f16, pre-scaled by
// 1/sqrt(K), in WMMA-B-fragment swizzled layout:
//   tile (kt,nt) covers k in [kt*32,kt*32+32), n in [nt*16,nt*16+16)
//   within a tile: 512 f16 = 32 lanes x 16 elems; lane holds column n=lane&15,
//   k_local = (lane>>4)*8 + elem (elem<8), 16 + (lane>>4)*8 + elem-8 (elem>=8)
// ---------------------------------------------------------------------------
__global__ void k_pack(const float* __restrict__ W, int K, int Nn, int Kp, int Np,
                       _Float16* __restrict__ out)
{
    int idx = blockIdx.x * blockDim.x + threadIdx.x;
    if (idx >= Kp * Np) return;
    int tile = idx >> 9;
    int lane = (idx >> 4) & 31;
    int el   = idx & 15;
    int ntiles = Np >> 4;
    int kt = tile / ntiles, nt = tile - kt * ntiles;
    int k = kt * 32 + ((lane >> 4) << 3) + (el & 7) + ((el >> 3) << 4);
    int n = nt * 16 + (lane & 15);
    float v = 0.f;
    if (k < K && n < Nn) v = W[(size_t)k * Nn + n] / sqrtf((float)K);
    out[idx] = (_Float16)v;
}

// A fragment 16x32 (f16): lane holds row m=lane&15,
// K = khb..khb+7 (v16h elems 0..7) and 16+khb..16+khb+7 (elems 8..15), khb = (lane>>4)*8
__device__ __forceinline__ v16h load_a_frag(const _Float16* __restrict__ p)
{
    v8h lo = *(const v8h*)(p);
    v8h hi = *(const v8h*)(p + 16);
    v16h a;
#pragma unroll
    for (int i = 0; i < 8; ++i) { a[i] = lo[i]; a[8 + i] = hi[i]; }
    return a;
}

#define WMMA_F16(A, B, C) __builtin_amdgcn_wmma_f32_16x16x32_f16( \
        false, (A), false, (B), (short)0, (C), false, false)

// ---------------------------------------------------------------------------
// WMMA GEMM: Out[M x Np] = act(A[M x Kp] @ Wsw), f16 in/out, f32 accumulate.
// One wave computes a 32(M) x 64(N) tile. K loop is unrolled 2x with two
// independent fragment register sets (structural double-buffering: no
// rotation movs), 16 WMMAs per iteration, loads overlapped with matrix ops.
// Requires Kp % 64 == 0 (Kp is 64 or 320 here).
// grid.x = M/32, grid.y = Np/64, block = 32 (one wave; EXEC all ones).
// ---------------------------------------------------------------------------
__global__ __launch_bounds__(32)
void k_gemm(const _Float16* __restrict__ A, int lda,
            const _Float16* __restrict__ Wsw, int Kp, int Np,
            _Float16* __restrict__ Out, int ldo, int act)
{
    int lane = threadIdx.x;
    int m0 = blockIdx.x << 5;
    int n0 = blockIdx.y << 6;
    int ntiles = Np >> 4;

    v8f acc[8];
#pragma unroll
    for (int t = 0; t < 8; ++t)
#pragma unroll
        for (int r = 0; r < 8; ++r) acc[t][r] = 0.f;

    int mrow = lane & 15;
    int khb  = (lane >> 4) << 3;            // 0 or 8
    const _Float16* arow0 = A + (size_t)(m0 + mrow) * lda + khb;
    const _Float16* arow1 = arow0 + (size_t)16 * lda;
    const _Float16* wt = Wsw + (((size_t)(n0 >> 4)) << 9) + ((size_t)lane << 4);
    const size_t wstep = (size_t)ntiles << 9;   // f16 elems per K-step of 32

    // prologue: fragment set 0 for k = 0
    v16h a0 = load_a_frag(arow0);
    v16h a1 = load_a_frag(arow1);
    v16h b0 = *(const v16h*)(wt);
    v16h b1 = *(const v16h*)(wt + 512);
    v16h b2 = *(const v16h*)(wt + 1024);
    v16h b3 = *(const v16h*)(wt + 1536);

    int k0 = 0;
    while (true) {
        // ---- load fragment set 1 for k0+32 (always valid: Kp%64==0) ----
        int kmid = k0 + 32;
        v16h c0 = load_a_frag(arow0 + kmid);
        v16h c1 = load_a_frag(arow1 + kmid);
        const _Float16* wtm = wt + (size_t)(kmid >> 5) * wstep;
        v16h d0 = *(const v16h*)(wtm);
        v16h d1 = *(const v16h*)(wtm + 512);
        v16h d2 = *(const v16h*)(wtm + 1024);
        v16h d3 = *(const v16h*)(wtm + 1536);

        // ---- WMMAs on set 0 ----
        acc[0] = WMMA_F16(a0, b0, acc[0]);
        acc[1] = WMMA_F16(a0, b1, acc[1]);
        acc[2] = WMMA_F16(a0, b2, acc[2]);
        acc[3] = WMMA_F16(a0, b3, acc[3]);
        acc[4] = WMMA_F16(a1, b0, acc[4]);
        acc[5] = WMMA_F16(a1, b1, acc[5]);
        acc[6] = WMMA_F16(a1, b2, acc[6]);
        acc[7] = WMMA_F16(a1, b3, acc[7]);

        // ---- load fragment set 0 for k0+64 (clamped: redundant on last) ----
        int knext = k0 + 64;
        bool more = (knext < Kp);
        int ksafe = more ? knext : 0;
        a0 = load_a_frag(arow0 + ksafe);
        a1 = load_a_frag(arow1 + ksafe);
        const _Float16* wtn = wt + (size_t)(ksafe >> 5) * wstep;
        b0 = *(const v16h*)(wtn);
        b1 = *(const v16h*)(wtn + 512);
        b2 = *(const v16h*)(wtn + 1024);
        b3 = *(const v16h*)(wtn + 1536);
        __builtin_prefetch(wt + (size_t)(ksafe >> 5) * wstep + wstep, 0, 1);

        // ---- WMMAs on set 1 ----
        acc[0] = WMMA_F16(c0, d0, acc[0]);
        acc[1] = WMMA_F16(c0, d1, acc[1]);
        acc[2] = WMMA_F16(c0, d2, acc[2]);
        acc[3] = WMMA_F16(c0, d3, acc[3]);
        acc[4] = WMMA_F16(c1, d0, acc[4]);
        acc[5] = WMMA_F16(c1, d1, acc[5]);
        acc[6] = WMMA_F16(c1, d2, acc[6]);
        acc[7] = WMMA_F16(c1, d3, acc[7]);

        if (!more) break;
        k0 = knext;
    }

    int mb = (lane >> 4) << 3;
    int nl = lane & 15;
#pragma unroll
    for (int h = 0; h < 2; ++h)
#pragma unroll
        for (int t = 0; t < 4; ++t)
#pragma unroll
            for (int rr = 0; rr < 8; ++rr) {
                float x = acc[h * 4 + t][rr];
                if (act) x = swish_f(x);
                Out[(size_t)(m0 + h * 16 + mb + rr) * ldo + n0 + t * 16 + nl] = (_Float16)x;
            }
}

// ---------------------------------------------------------------------------
// Edge geometry: r and component-normalized Y1 (sqrt(3)*v/r/sqrt(20))
// ---------------------------------------------------------------------------
__global__ void k_geom(const float* __restrict__ pos, const int* __restrict__ ei,
                       int E, float* __restrict__ r, float* __restrict__ sh1)
{
    int e = blockIdx.x * blockDim.x + threadIdx.x;
    if (e >= E) return;
    int a = ei[e], b = ei[E + e];
    float dx = pos[a * 3]     - pos[b * 3];
    float dy = pos[a * 3 + 1] - pos[b * 3 + 1];
    float dz = pos[a * 3 + 2] - pos[b * 3 + 2];
    float rr = sqrtf(dx * dx + dy * dy + dz * dz);
    r[e] = rr;
    float s = 1.7320508075688772f / rr * SH0c;   // sqrt(3)/r/sqrt(20)
    sh1[e * 3]     = dx * s;
    sh1[e * 3 + 1] = dy * s;
    sh1[e * 3 + 2] = dz * s;
}

// Gaussian basis for an edge chunk -> f16 [Mc x 64] (cols>=40 and rows>=E zero)
__global__ void k_basis(const float* __restrict__ r, int E, int base, int Mc,
                        _Float16* __restrict__ A)
{
    int idx = blockIdx.x * blockDim.x + threadIdx.x;
    if (idx >= Mc * 64) return;
    int le = idx >> 6, j = idx & 63;
    int e = base + le;
    float v = 0.f;
    if (e < E && j < 40) {
        const float step = 10.f / 39.f;
        float d = (r[e] - (float)j * step) / step;
        v = __expf(-d * d);
    }
    A[(size_t)le * 64 + j] = (_Float16)v;
}

__global__ void k_embed(const int* __restrict__ z, const float* __restrict__ emb,
                        float* __restrict__ h, int N)
{
    int idx = blockIdx.x * blockDim.x + threadIdx.x;
    if (idx >= N * 50) return;
    int n = idx / 50, i = idx - n * 50;
    h[idx] = emb[z[n] * 50 + i];
}

// ---------------- layer 1 ----------------
__global__ void k_init1(int N, const float* __restrict__ h, const float* __restrict__ si1,
                        float* __restrict__ hs1, float* __restrict__ hv1)
{
    int idx = blockIdx.x * blockDim.x + threadIdx.x;
    if (idx >= N * 60) return;
    int n = idx / 60, o = idx - n * 60;
    float s = 0.f;
    for (int i = 0; i < 50; ++i) s += h[n * 50 + i] * si1[i * 60 + o];
    hs1[idx] = SQH * RS50 * s;
    if (o < 30) hv1[n * 30 + o] = 0.f;
}

__global__ void k_msg1(const int* __restrict__ ei, int E, int base, int Mc,
                       const _Float16* __restrict__ wbuf, const float* __restrict__ h,
                       const float* __restrict__ sh1,
                       float* __restrict__ hs1, float* __restrict__ hv1)
{
    long idx = (long)blockIdx.x * blockDim.x + threadIdx.x;
    int le = (int)(idx / 70), slot = (int)(idx % 70);
    if (le >= Mc) return;
    int e = base + le;
    if (e >= E) return;
    int row = ei[e], col = ei[E + e];
    const _Float16* w = wbuf + (size_t)le * 3520;
    const float* xj = h + (size_t)row * 50;
    if (slot < 60) {
        float s = 0.f;
        for (int i = 0; i < 50; ++i) s += xj[i] * (float)w[i * 60 + slot];
        atomicAdd(&hs1[(size_t)col * 60 + slot], s * (SQH * SH0c * RS50));
    } else {
        int o = slot - 60;
        float s = 0.f;
        for (int i = 0; i < 50; ++i) s += xj[i] * (float)w[3000 + i * 10 + o];
        s *= RS50;
        for (int c = 0; c < 3; ++c)
            atomicAdd(&hv1[((size_t)col * 10 + o) * 3 + c], s * sh1[(size_t)e * 3 + c]);
    }
}

__global__ void k_gate1(int N, const float* __restrict__ hs1, const float* __restrict__ hv1,
                        float* __restrict__ xs1, float* __restrict__ vo1)
{
    int n = blockIdx.x * blockDim.x + threadIdx.x;
    if (n >= N) return;
    for (int i = 0; i < 50; ++i) xs1[n * 50 + i] = swish_f(hs1[n * 60 + i]);
    for (int j = 0; j < 10; ++j) {
        float g = sigm_f(hs1[n * 60 + 50 + j]);
        for (int c = 0; c < 3; ++c)
            vo1[n * 30 + j * 3 + c] = hv1[n * 30 + j * 3 + c] * g;
    }
}

// ---------------- layer 2 ----------------
__global__ void k_init2(int N, const float* __restrict__ xs1, const float* __restrict__ vo1,
                        const float* __restrict__ si2s, const float* __restrict__ si2v,
                        float* __restrict__ hs2, float* __restrict__ hvo,
                        float* __restrict__ hve)
{
    int idx = blockIdx.x * blockDim.x + threadIdx.x;
    if (idx >= N * 100) return;
    int n = idx / 100, q = idx - n * 100;
    if (q < 70) {
        float s = 0.f;
        for (int i = 0; i < 50; ++i) s += xs1[n * 50 + i] * si2s[i * 70 + q];
        hs2[n * 70 + q] = SQH * RS50 * s;
    } else {
        int t = q - 70, o = t / 3, c = t - o * 3;
        float s = 0.f;
        for (int i = 0; i < 10; ++i) s += vo1[n * 30 + i * 3 + c] * si2v[i * 10 + o];
        hvo[n * 30 + o * 3 + c] = SQH * RS10 * s;
        hve[n * 30 + t] = 0.f;
    }
}

__global__ void k_msg2(const int* __restrict__ ei, int E, int base, int Mc,
                       const _Float16* __restrict__ wbuf,
                       const float* __restrict__ xs1, const float* __restrict__ vo1,
                       const float* __restrict__ sh1,
                       float* __restrict__ hs2, float* __restrict__ hvo,
                       float* __restrict__ hve)
{
    long idx = (long)blockIdx.x * blockDim.x + threadIdx.x;
    int le = (int)(idx / 130), slot = (int)(idx % 130);
    if (le >= Mc) return;
    int e = base + le;
    if (e >= E) return;
    int row = ei[e], col = ei[E + e];
    const _Float16* w = wbuf + (size_t)le * 4928;
    const float* xj = xs1 + (size_t)row * 50;
    const float* vj = vo1 + (size_t)row * 30;
    float s0 = sh1[(size_t)e * 3], s1 = sh1[(size_t)e * 3 + 1], s2 = sh1[(size_t)e * 3 + 2];
    if (slot < 70) {
        float a = 0.f;
        for (int i = 0; i < 50; ++i) a += xj[i] * (float)w[i * 70 + slot];
        float b = 0.f;
        for (int i = 0; i < 10; ++i) {
            float dv = vj[i * 3] * s0 + vj[i * 3 + 1] * s1 + vj[i * 3 + 2] * s2;
            b += dv * (float)w[3500 + i * 70 + slot];
        }
        atomicAdd(&hs2[(size_t)col * 70 + slot], SQH * RS60 * (a * SH0c + b * RS3));
    } else if (slot < 100) {
        int q = slot - 70, o = q / 3, c = q - o * 3;
        float sc = (c == 0) ? s0 : ((c == 1) ? s1 : s2);
        float a = 0.f;
        for (int i = 0; i < 50; ++i) a += xj[i] * (float)w[4200 + i * 10 + o];
        float b = 0.f;
        for (int i = 0; i < 10; ++i) b += vj[i * 3 + c] * (float)w[4700 + i * 10 + o];
        atomicAdd(&hvo[((size_t)col * 10 + o) * 3 + c], SQH * RS60 * (a * sc + b * SH0c));
    } else {
        int q = slot - 100, o = q / 3, c = q - o * 3;
        int c1 = (c + 1) % 3, c2 = (c + 2) % 3;
        float sb1 = (c1 == 0) ? s0 : ((c1 == 1) ? s1 : s2);
        float sb2 = (c2 == 0) ? s0 : ((c2 == 1) ? s1 : s2);
        float b = 0.f;
        for (int i = 0; i < 10; ++i) {
            float cr = vj[i * 3 + c1] * sb2 - vj[i * 3 + c2] * sb1;
            b += cr * (float)w[4800 + i * 10 + o];
        }
        atomicAdd(&hve[((size_t)col * 10 + o) * 3 + c], RS2 * RS10 * b);
    }
}

__global__ void k_gate2(int N, const float* __restrict__ hs2, const float* __restrict__ hvo,
                        const float* __restrict__ hve,
                        float* __restrict__ xs2, float* __restrict__ vo2,
                        float* __restrict__ vev)
{
    int n = blockIdx.x * blockDim.x + threadIdx.x;
    if (n >= N) return;
    for (int i = 0; i < 50; ++i) xs2[n * 50 + i] = swish_f(hs2[n * 70 + i]);
    for (int j = 0; j < 10; ++j) {
        float go = sigm_f(hs2[n * 70 + 50 + j]);
        float ge = sigm_f(hs2[n * 70 + 60 + j]);
        for (int c = 0; c < 3; ++c) {
            vo2[n * 30 + j * 3 + c] = hvo[n * 30 + j * 3 + c] * go;
            vev[n * 30 + j * 3 + c] = hve[n * 30 + j * 3 + c] * ge;
        }
    }
}

// ---------------- final conv + readout ----------------
__global__ void k_init3(int N, const float* __restrict__ xs2, const float* __restrict__ si3,
                        float* __restrict__ he, float* __restrict__ ho)
{
    int n = blockIdx.x * blockDim.x + threadIdx.x;
    if (n >= N) return;
    float s = 0.f;
    for (int i = 0; i < 50; ++i) s += xs2[n * 50 + i] * si3[i];
    he[n] = SQH * RS50 * s;
    ho[n] = 0.f;
}

__global__ void k_msg3(const int* __restrict__ ei, int E, int base, int Mc,
                       const _Float16* __restrict__ wbuf,
                       const float* __restrict__ xs2, const float* __restrict__ vo2,
                       const float* __restrict__ vev, const float* __restrict__ sh1,
                       float* __restrict__ he, float* __restrict__ ho)
{
    int le = blockIdx.x * blockDim.x + threadIdx.x;
    if (le >= Mc) return;
    int e = base + le;
    if (e >= E) return;
    int row = ei[e], col = ei[E + e];
    const _Float16* w = wbuf + (size_t)le * 128;
    const float* xj = xs2 + (size_t)row * 50;
    const float* vo = vo2 + (size_t)row * 30;
    const float* ve = vev + (size_t)row * 30;
    float s0 = sh1[(size_t)e * 3], s1 = sh1[(size_t)e * 3 + 1], s2 = sh1[(size_t)e * 3 + 2];
    float a = 0.f;
    for (int i = 0; i < 50; ++i) a += xj[i] * (float)w[i];
    float b = 0.f, d = 0.f;
    for (int i = 0; i < 10; ++i) {
        b += (vo[i * 3] * s0 + vo[i * 3 + 1] * s1 + vo[i * 3 + 2] * s2) * (float)w[50 + i];
        d += (ve[i * 3] * s0 + ve[i * 3 + 1] * s1 + ve[i * 3 + 2] * s2) * (float)w[60 + i];
    }
    atomicAdd(&he[col], SQH * RS60 * (a * SH0c + b * RS3));
    atomicAdd(&ho[col], RS3 * RS10 * d);
}

__global__ void k_readout(const int* __restrict__ batch, const float* __restrict__ he,
                          const float* __restrict__ ho, float* __restrict__ out, int N)
{
    int g = blockIdx.x * blockDim.x + threadIdx.x;
    if (g >= 64) return;
    float s = 0.f;
    for (int n = 0; n < N; ++n)
        if (batch[n] == g) s += he[n] + 0.5f * ho[n] * ho[n];
    out[g] = s;
}

// ---------------------------------------------------------------------------
extern "C" void kernel_launch(void* const* d_in, const int* in_sizes, int n_in,
                              void* d_out, int out_size, void* d_ws, size_t ws_size,
                              hipStream_t stream)
{
    const int N = 1280;
    const int* z      = (const int*)d_in[0];
    const float* pos  = (const float*)d_in[1];
    const int* batch  = (const int*)d_in[2];
    const int* ei     = (const int*)d_in[3];
    const float* emb  = (const float*)d_in[4];
    int E = in_sizes[3] / 2;
    int Epad = (E + 31) & ~31;          // M-tile is 32 rows

    static const int dims[7] = {40, 300, 300, 300, 300, 50, 0};
    static const int NW[3]   = {3500, 4900, 70};
    static const int NWp[3]  = {3520, 4928, 128};
    static const int KP[6]   = {64, 320, 320, 320, 320, 64};
    static const int NPbase[6] = {320, 320, 320, 320, 64, 0};

    const float* radW[3][6];
    const float *si1, *si2s, *si2v, *si3;
    if (n_in >= 27) {
        for (int m = 0; m < 3; ++m)
            for (int l = 0; l < 6; ++l) radW[m][l] = (const float*)d_in[5 + m * 6 + l];
        si1  = (const float*)d_in[23];
        si2s = (const float*)d_in[24];
        si2v = (const float*)d_in[25];
        si3  = (const float*)d_in[26];
    } else {  // pytree tuples flattened as single concatenated blobs
        for (int m = 0; m < 3; ++m) {
            const float* p = (const float*)d_in[5 + m];
            for (int l = 0; l < 6; ++l) {
                radW[m][l] = p;
                int K = dims[l], Nn = (l == 5) ? NW[m] : dims[l + 1];
                p += (size_t)K * Nn;
            }
        }
        si1  = (const float*)d_in[8];
        si2s = (const float*)d_in[9];
        si2v = (const float*)d_in[10];
        si3  = (const float*)d_in[11];
    }

    // ---- workspace bump allocator (256B aligned) ----
    char* ws = (char*)d_ws;
    size_t off = 0;
    auto alloc = [&](size_t bytes) -> void* {
        void* p = ws + off;
        off = (off + bytes + 255) & ~(size_t)255;
        return p;
    };
    float* rbuf  = (float*)alloc((size_t)E * 4);
    float* sh1b  = (float*)alloc((size_t)E * 12);
    float* hbuf  = (float*)alloc((size_t)N * 50 * 4);
    float* hs1   = (float*)alloc((size_t)N * 60 * 4);
    float* hv1   = (float*)alloc((size_t)N * 30 * 4);
    float* xs1   = (float*)alloc((size_t)N * 50 * 4);
    float* vo1   = (float*)alloc((size_t)N * 30 * 4);
    float* hs2   = (float*)alloc((size_t)N * 70 * 4);
    float* hvo   = (float*)alloc((size_t)N * 30 * 4);
    float* hve   = (float*)alloc((size_t)N * 30 * 4);
    float* xs2   = (float*)alloc((size_t)N * 50 * 4);
    float* vo2   = (float*)alloc((size_t)N * 30 * 4);
    float* vev   = (float*)alloc((size_t)N * 30 * 4);
    float* he    = (float*)alloc((size_t)N * 4);
    float* ho    = (float*)alloc((size_t)N * 4);

    _Float16* wp[3][6];
    for (int m = 0; m < 3; ++m)
        for (int l = 0; l < 6; ++l) {
            int Np = (l == 5) ? NWp[m] : NPbase[l];
            wp[m][l] = (_Float16*)alloc((size_t)KP[l] * Np * 2);
        }

    // edge-chunk size from remaining workspace (multiple of 32 rows)
    const size_t per_edge = (size_t)(320 + 320 + 4928) * 2;
    size_t remain = (ws_size > off + 1024) ? (ws_size - off - 1024) : 0;
    long ecl = (long)(remain / per_edge);
    int EC = (int)(ecl & ~31L);
    if (EC > Epad) EC = Epad;
    if (EC < 32) EC = 32;
    _Float16* actP = (_Float16*)alloc((size_t)EC * 320 * 2);
    _Float16* actQ = (_Float16*)alloc((size_t)EC * 320 * 2);
    _Float16* wbuf = (_Float16*)alloc((size_t)EC * 4928 * 2);
    int nch = (Epad + EC - 1) / EC;

    // ---- pack all MLP weights (f32 -> swizzled f16, scale folded in) ----
    for (int m = 0; m < 3; ++m)
        for (int l = 0; l < 6; ++l) {
            int Np = (l == 5) ? NWp[m] : NPbase[l];
            int Nn = (l == 5) ? NW[m] : dims[l + 1];
            int total = KP[l] * Np;
            k_pack<<<(total + 255) / 256, 256, 0, stream>>>(radW[m][l], dims[l], Nn,
                                                            KP[l], Np, wp[m][l]);
        }

    k_geom<<<(E + 255) / 256, 256, 0, stream>>>(pos, ei, E, rbuf, sh1b);
    k_embed<<<(N * 50 + 255) / 256, 256, 0, stream>>>(z, emb, hbuf, N);

    auto run_mlp = [&](int m, int Mc) {
        int mt = Mc / 32;
        k_gemm<<<dim3(mt, 5), 32, 0, stream>>>(actP, 64,  wp[m][0], 64,  320, actQ, 320, 1);
        k_gemm<<<dim3(mt, 5), 32, 0, stream>>>(actQ, 320, wp[m][1], 320, 320, actP, 320, 1);
        k_gemm<<<dim3(mt, 5), 32, 0, stream>>>(actP, 320, wp[m][2], 320, 320, actQ, 320, 1);
        k_gemm<<<dim3(mt, 5), 32, 0, stream>>>(actQ, 320, wp[m][3], 320, 320, actP, 320, 1);
        k_gemm<<<dim3(mt, 1), 32, 0, stream>>>(actP, 320, wp[m][4], 320, 64,  actQ, 64,  1);
        k_gemm<<<dim3(mt, NWp[m] / 64), 32, 0, stream>>>(actQ, 64, wp[m][5], 64, NWp[m],
                                                         wbuf, NWp[m], 0);
    };

    // ---- layer 1 ----
    k_init1<<<(N * 60 + 255) / 256, 256, 0, stream>>>(N, hbuf, si1, hs1, hv1);
    for (int ci = 0; ci < nch; ++ci) {
        int base = ci * EC;
        int Mc = Epad - base; if (Mc > EC) Mc = EC;
        k_basis<<<(Mc * 64 + 255) / 256, 256, 0, stream>>>(rbuf, E, base, Mc, actP);
        run_mlp(0, Mc);
        long th = (long)Mc * 70;
        k_msg1<<<(unsigned)((th + 255) / 256), 256, 0, stream>>>(ei, E, base, Mc, wbuf,
                                                                 hbuf, sh1b, hs1, hv1);
    }
    k_gate1<<<(N + 255) / 256, 256, 0, stream>>>(N, hs1, hv1, xs1, vo1);

    // ---- layer 2 ----
    k_init2<<<(N * 100 + 255) / 256, 256, 0, stream>>>(N, xs1, vo1, si2s, si2v, hs2, hvo, hve);
    for (int ci = 0; ci < nch; ++ci) {
        int base = ci * EC;
        int Mc = Epad - base; if (Mc > EC) Mc = EC;
        k_basis<<<(Mc * 64 + 255) / 256, 256, 0, stream>>>(rbuf, E, base, Mc, actP);
        run_mlp(1, Mc);
        long th = (long)Mc * 130;
        k_msg2<<<(unsigned)((th + 255) / 256), 256, 0, stream>>>(ei, E, base, Mc, wbuf,
                                                                 xs1, vo1, sh1b, hs2, hvo, hve);
    }
    k_gate2<<<(N + 255) / 256, 256, 0, stream>>>(N, hs2, hvo, hve, xs2, vo2, vev);

    // ---- final conv ----
    k_init3<<<(N + 255) / 256, 256, 0, stream>>>(N, xs2, si3, he, ho);
    for (int ci = 0; ci < nch; ++ci) {
        int base = ci * EC;
        int Mc = Epad - base; if (Mc > EC) Mc = EC;
        k_basis<<<(Mc * 64 + 255) / 256, 256, 0, stream>>>(rbuf, E, base, Mc, actP);
        run_mlp(2, Mc);
        k_msg3<<<(Mc + 255) / 256, 256, 0, stream>>>(ei, E, base, Mc, wbuf,
                                                     xs2, vo2, vev, sh1b, he, ho);
    }

    k_readout<<<1, 64, 0, stream>>>(batch, he, ho, (float*)d_out, N);
}
